// MoHBertSelfAttention_32633161515472
// MI455X (gfx1250) — matrix operand order repaired and checked by reading, not compiled, not run on hardware
//
#include <hip/hip_runtime.h>
#include <hip/hip_bf16.h>
#include <math.h>

typedef __bf16         v16bf __attribute__((ext_vector_type(16)));
typedef float          v8f   __attribute__((ext_vector_type(8)));
typedef float          f32x4 __attribute__((ext_vector_type(4)));
typedef unsigned int   u32x4 __attribute__((ext_vector_type(4)));
typedef unsigned short u16x4 __attribute__((ext_vector_type(4)));

#define BSZ   4
#define TSZ   1024
#define HIDSZ 768
#define NH    12
#define HD    64
#define DROWS 2064            // 2047 real dist rows, zero-padded to cover the band
#define QE_BYTES 33280        // 16 x 1040 bf16 (probs 16x1024 bf16 overlays this later)
#define S_BYTES  65536        // 16 x 1024 f32
#define WAVE_LDS (QE_BYTES + S_BYTES)

__device__ __forceinline__ unsigned short f2bf(float f) {
  unsigned u = __builtin_bit_cast(unsigned, f);
  u += 0x7fffu + ((u >> 16) & 1u);          // round-to-nearest-even
  return (unsigned short)(u >> 16);
}
__device__ __forceinline__ float bf2f(unsigned short s) {
  unsigned u = ((unsigned)s) << 16;
  return __builtin_bit_cast(float, u);
}

struct U4x2 { u32x4 lo, hi; };
__device__ __forceinline__ v16bf make_frag(u32x4 lo, u32x4 hi) {
  U4x2 t; t.lo = lo; t.hi = hi;
  return __builtin_bit_cast(v16bf, t);
}
#define WMMA_BF16(A, B, C) \
  __builtin_amdgcn_wmma_f32_16x16x32_bf16(false, (A), false, (B), (short)0, (C), false, false)
#define V8F_ZERO {0.f,0.f,0.f,0.f,0.f,0.f,0.f,0.f}

// ---------------------------------------------------------------- converters
__global__ void cvt_bf16_kernel(const float* __restrict__ src,
                                unsigned short* __restrict__ dst, int n) {
  for (int i = blockIdx.x * blockDim.x + threadIdx.x; i < n; i += gridDim.x * blockDim.x)
    dst[i] = f2bf(src[i]);
}
__global__ void cvt_dist_kernel(const float* __restrict__ src,
                                unsigned short* __restrict__ dst) {
  const int n = DROWS * HD, real = 2047 * HD;
  for (int i = blockIdx.x * blockDim.x + threadIdx.x; i < n; i += gridDim.x * blockDim.x)
    dst[i] = (i < real) ? f2bf(src[i]) : (unsigned short)0;
}

// ---------------------------------------------------------------- QKV GEMM
// Per-matrix body templated on MAT so each weight pointer is used directly
// (keeps loads in the GLOBAL address space -> global_load_b128 + saddr,
// no flat_load / loadcnt+dscnt double-drain). B operand is double-buffered.
template <int MAT>
__device__ __forceinline__ void qkv_mat(
    const v16bf (&a)[24], const unsigned short* __restrict__ W,
    const float* __restrict__ bias, int tr0, int half, int l16, int wave,
    unsigned short* __restrict__ Qb, unsigned short* __restrict__ Kb,
    unsigned short* __restrict__ Vt) {
  for (int nt = wave; nt < 48; nt += 4) {
    const int n = nt * 16 + l16;                         // out-feature (B col)
    const unsigned short* wrow = W + (size_t)n * HIDSZ + half * 16;
    v8f acc = V8F_ZERO;
    u32x4 bl = *(const u32x4*)(wrow);
    u32x4 bh = *(const u32x4*)(wrow + 8);
#pragma unroll
    for (int kc = 0; kc < 24; ++kc) {                    // K = 768 = 24 x 32
      const v16bf bm = make_frag(bl, bh);
      if (kc < 23) {                                     // prefetch next chunk
        bl = *(const u32x4*)(wrow + (kc + 1) * 32);
        bh = *(const u32x4*)(wrow + (kc + 1) * 32 + 8);
      }
      acc = WMMA_BF16(a[kc], bm, acc);
    }
    const float bv = bias[n];
    const int hh = n >> 6, d = n & 63;
#pragma unroll
    for (int g = 0; g < 8; ++g) {
      const int tr = tr0 + g + half * 8;                 // global token row
      const int b = tr >> 10, tt = tr & 1023;
      const unsigned short o = f2bf(acc[g] + bv);
      const size_t bh2 = (size_t)b * NH + hh;
      if (MAT == 0)      Qb[(bh2 * TSZ + tt) * HD + d] = o;
      else if (MAT == 1) Kb[(bh2 * TSZ + tt) * HD + d] = o;
      else               Vt[(bh2 * HD + d) * TSZ + tt] = o;   // transposed
    }
  }
}

__global__ __launch_bounds__(128) void qkv_kernel(
    const unsigned short* __restrict__ xb,
    const unsigned short* __restrict__ wq, const unsigned short* __restrict__ wk,
    const unsigned short* __restrict__ wv,
    const float* __restrict__ bq, const float* __restrict__ bk,
    const float* __restrict__ bv,
    unsigned short* __restrict__ Qb, unsigned short* __restrict__ Kb,
    unsigned short* __restrict__ Vt) {
  __shared__ unsigned short xt[16 * HIDSZ];
  const int tid = threadIdx.x;
  const int tr0 = blockIdx.x * 16;                       // token-row tile base
  { // cooperative global->LDS stage (16*768*2B = 1536 x b128)
    const u32x4* src = (const u32x4*)(xb + (size_t)tr0 * HIDSZ);
    u32x4* dst = (u32x4*)xt;
    for (int i = tid; i < 1536; i += 128) dst[i] = src[i];
  }
  __syncthreads();

  const int lane = tid & 31, wave = tid >> 5;
  const int half = lane >> 4, l16 = lane & 15;

  // A-fragments for the whole K=768 reduction, loaded from LDS once per wave
  v16bf a[24];
#pragma unroll
  for (int kc = 0; kc < 24; ++kc) {
    const unsigned short* arow = xt + l16 * HIDSZ + kc * 32 + half * 16;
    a[kc] = make_frag(*(const u32x4*)arow, *(const u32x4*)(arow + 8));
  }

  qkv_mat<0>(a, wq, bq, tr0, half, l16, wave, Qb, Kb, Vt);
  qkv_mat<1>(a, wk, bk, tr0, half, l16, wave, Qb, Kb, Vt);
  qkv_mat<2>(a, wv, bv, tr0, half, l16, wave, Qb, Kb, Vt);
}

// ---------------------------------------------------------------- MoH gating
__global__ __launch_bounds__(128) void gate_kernel(
    const float* __restrict__ x, const float* __restrict__ Wr,
    const float* __restrict__ br, const float* __restrict__ Wa,
    float* __restrict__ wbuf) {
  const int lane = threadIdx.x & 31;
  const int t = blockIdx.x * 4 + (threadIdx.x >> 5);
  const float* xr = x + (size_t)t * HIDSZ;
  float a0 = 0, a1 = 0, a2 = 0, a3 = 0, a4 = 0, a5 = 0;
  for (int k = lane; k < HIDSZ; k += 32) {
    const float xv = xr[k];
    a0 += xv * Wr[0 * HIDSZ + k];
    a1 += xv * Wr[1 * HIDSZ + k];
    a2 += xv * Wr[2 * HIDSZ + k];
    a3 += xv * Wr[3 * HIDSZ + k];
    a4 += xv * Wa[0 * HIDSZ + k];
    a5 += xv * Wa[1 * HIDSZ + k];
  }
  for (int m = 16; m; m >>= 1) {
    a0 += __shfl_xor(a0, m, 32); a1 += __shfl_xor(a1, m, 32);
    a2 += __shfl_xor(a2, m, 32); a3 += __shfl_xor(a3, m, 32);
    a4 += __shfl_xor(a4, m, 32); a5 += __shfl_xor(a5, m, 32);
  }
  if (lane == 0) {
    float g[4];                                         // relu(clip(v,-1,10)) == clamp(v,0,10)
    g[0] = fminf(fmaxf(a0 + br[0], 0.f), 10.f);
    g[1] = fminf(fmaxf(a1 + br[1], 0.f), 10.f);
    g[2] = fminf(fmaxf(a2 + br[2], 0.f), 10.f);
    g[3] = fminf(fmaxf(a3 + br[3], 0.f), 10.f);
    const float c0 = fminf(fmaxf(a4, -1.f), 1.f);
    const float c1 = fminf(fmaxf(a5, -1.f), 1.f);
    const float mx = fmaxf(c0, c1);
    const float e0 = __expf(c0 - mx), e1 = __expf(c1 - mx);
    const float al0 = e0 / (e0 + e1), al1 = e1 / (e0 + e1);
    const float m1 = fmaxf(fmaxf(g[0], g[1]), fmaxf(g[2], g[3]));
    float s2 = -3.4e38f; bool used = false;             // 2nd-largest (ties -> equal)
    for (int i = 0; i < 4; ++i) {
      if (!used && g[i] == m1) used = true;
      else s2 = fmaxf(s2, g[i]);
    }
    float gg[4], sum = 0.f;
    for (int i = 0; i < 4; ++i) { gg[i] = (g[i] >= s2) ? g[i] : 0.f; sum += gg[i]; }
    const float invs = 1.f / (sum + 1e-9f);
    float* wp = wbuf + (size_t)t * 12;
    for (int i = 0; i < 8; ++i) wp[i] = al0;            // shared heads
    for (int i = 0; i < 4; ++i) wp[8 + i] = al1 * gg[i] * invs;  // routed heads
  }
}

// ---------------------------------------------------------------- attention
// One wave per 16-row l-tile. (1) QE band GEMM -> LDS, (2) QK^T + gathered
// rel -> S(f32 LDS), (3) softmax (2 lanes/row), probs bf16 overlay QE,
// (4) P*V GEMM, epilogue applies 1/rowsum * MoH head weight.
// All global B operands double-buffered across tiles.
struct BPair { u32x4 l0, h0, l1, h1; };                 // two K=32 fragments
__device__ __forceinline__ BPair load_bpair(const unsigned short* row) {
  BPair p;
  p.l0 = *(const u32x4*)(row);       p.h0 = *(const u32x4*)(row + 8);
  p.l1 = *(const u32x4*)(row + 32);  p.h1 = *(const u32x4*)(row + 40);
  return p;
}

__global__ __launch_bounds__(64) void attn_kernel(
    const unsigned short* __restrict__ Qb, const unsigned short* __restrict__ Kb,
    const unsigned short* __restrict__ Vt, const unsigned short* __restrict__ distb,
    const float* __restrict__ wbuf, float* __restrict__ out) {
  extern __shared__ char smem[];
  const int lane = threadIdx.x & 31;
  const int wave = threadIdx.x >> 5;
  const int tile = blockIdx.x * 2 + wave;               // 0..3071
  const int bh = tile >> 6;                             // (b,h) pair
  const int L0 = (tile & 63) * 16;
  const int b = bh / NH, h = bh % NH;
  const int half = lane >> 4, l16 = lane & 15;

  char* base = smem + wave * WAVE_LDS;
  unsigned short* qe = (unsigned short*)base;           // 16x1040 bf16; later probs 16x1024
  float* S = (float*)(base + QE_BYTES);                 // 16x1024 f32

  // Q A-fragments (K=64 -> two K=32 chunks), resident all kernel
  const unsigned short* qrow = Qb + ((size_t)bh * TSZ + L0 + l16) * HD;
  v16bf aq[2];
#pragma unroll
  for (int c = 0; c < 2; ++c) {
    const unsigned short* p = qrow + c * 32 + half * 16;
    aq[c] = make_frag(*(const u32x4*)p, *(const u32x4*)(p + 8));
  }

  // Phase 1: QE[i,mm] = q[L0+i] . dist_emb[L0+mm], mm in [0,1040)  (band GEMM)
  {
    const unsigned short* dbase = distb + (size_t)(L0 + l16) * HD + half * 16;
    BPair cur = load_bpair(dbase);
    for (int mt = 0; mt < 65; ++mt) {
      const BPair use = cur;
      if (mt + 1 < 65) cur = load_bpair(dbase + (size_t)(mt + 1) * 16 * HD);
      v8f acc = V8F_ZERO;
      acc = WMMA_BF16(aq[0], make_frag(use.l0, use.h0), acc);
      acc = WMMA_BF16(aq[1], make_frag(use.l1, use.h1), acc);
#pragma unroll
      for (int g = 0; g < 8; ++g) {
        const int M = g + half * 8;
        qe[M * 1040 + mt * 16 + l16] = f2bf(acc[g]);
      }
    }
  }
  __syncthreads();

  // Phase 2: scores = (Q.K^T + rel) / sqrt(D) -> S
  {
    const unsigned short* kbase = Kb + ((size_t)bh * TSZ + l16) * HD + half * 16;
    BPair cur = load_bpair(kbase);
    for (int rt = 0; rt < 64; ++rt) {
      const BPair use = cur;
      if (rt + 1 < 64) cur = load_bpair(kbase + (size_t)(rt + 1) * 16 * HD);
      v8f acc = V8F_ZERO;
      acc = WMMA_BF16(aq[0], make_frag(use.l0, use.h0), acc);
      acc = WMMA_BF16(aq[1], make_frag(use.l1, use.h1), acc);
      const int coff = 1023 - rt * 16;
#pragma unroll
      for (int g = 0; g < 8; ++g) {
        const int M = g + half * 8;
        const int mm = M - l16 + coff;                  // band gather index
        S[M * 1024 + rt * 16 + l16] = (acc[g] + bf2f(qe[M * 1040 + mm])) * 0.125f;
      }
    }
  }
  __syncthreads();

  // Phase 3: softmax, 2 lanes per row; probs bf16 overlay the dead QE region
  {
    const int row = lane >> 1, part = lane & 1;
    const f32x4* sp = (const f32x4*)(S + row * 1024 + part * 512);
    float m = -3.4e38f;
    for (int i = 0; i < 128; ++i) {
      const f32x4 v = sp[i];
      m = fmaxf(m, fmaxf(fmaxf(v[0], v[1]), fmaxf(v[2], v[3])));
    }
    m = fmaxf(m, __shfl_xor(m, 1, 32));
    float sum = 0.f;
    u16x4* pp = (u16x4*)(qe + row * 1024 + part * 512);
    for (int i = 0; i < 128; ++i) {
      const f32x4 v = sp[i];
      const float e0 = __expf(v[0] - m), e1 = __expf(v[1] - m);
      const float e2 = __expf(v[2] - m), e3 = __expf(v[3] - m);
      sum += (e0 + e1) + (e2 + e3);
      u16x4 pv; pv[0] = f2bf(e0); pv[1] = f2bf(e1); pv[2] = f2bf(e2); pv[3] = f2bf(e3);
      pp[i] = pv;
    }
    sum += __shfl_xor(sum, 1, 32);
    if (part == 0) S[row] = 1.f / sum;                  // S dead: stash 1/rowsum
  }
  __syncthreads();

  // Phase 4: ctx = P.V, epilogue * (1/rowsum) * MoH head weight
  float inv8[8], wv8[8];
#pragma unroll
  for (int g = 0; g < 8; ++g) {
    const int M = g + half * 8;
    inv8[g] = S[M];
    wv8[g] = wbuf[((size_t)b * TSZ + L0 + M) * 12 + h];
  }
  for (int dt = 0; dt < 4; ++dt) {
    const unsigned short* vrow = Vt + ((size_t)bh * HD + dt * 16 + l16) * TSZ + half * 16;
    v8f acc = V8F_ZERO;
    u32x4 bl = *(const u32x4*)(vrow);
    u32x4 bhv = *(const u32x4*)(vrow + 8);
    for (int kc = 0; kc < 32; ++kc) {                   // K = 1024 = 32 x 32
      const v16bf bm = make_frag(bl, bhv);
      if (kc + 1 < 32) {                                // prefetch next chunk
        bl  = *(const u32x4*)(vrow + (kc + 1) * 32);
        bhv = *(const u32x4*)(vrow + (kc + 1) * 32 + 8);
      }
      const unsigned short* ap = qe + l16 * 1024 + kc * 32 + half * 16;
      const v16bf am = make_frag(*(const u32x4*)ap, *(const u32x4*)(ap + 8));
      acc = WMMA_BF16(am, bm, acc);
    }
#pragma unroll
    for (int g = 0; g < 8; ++g) {
      const int M = g + half * 8;
      const int t = L0 + M, d = dt * 16 + l16;
      out[((size_t)b * TSZ + t) * HIDSZ + h * HD + d] = acc[g] * inv8[g] * wv8[g];
    }
  }
}

// ---------------------------------------------------------------- launch
extern "C" void kernel_launch(void* const* d_in, const int* in_sizes, int n_in,
                              void* d_out, int out_size, void* d_ws, size_t ws_size,
                              hipStream_t stream) {
  const float* x    = (const float*)d_in[0];
  const float* Wq   = (const float*)d_in[1];
  const float* bq   = (const float*)d_in[2];
  const float* Wk   = (const float*)d_in[3];
  const float* bk   = (const float*)d_in[4];
  const float* Wv   = (const float*)d_in[5];
  const float* bv   = (const float*)d_in[6];
  const float* dist = (const float*)d_in[7];
  const float* Wr   = (const float*)d_in[8];
  const float* br   = (const float*)d_in[9];
  const float* Wa   = (const float*)d_in[10];
  float* out = (float*)d_out;

  char* ws = (char*)d_ws;
  size_t off = 0;
  unsigned short* xb   = (unsigned short*)(ws + off); off += (size_t)BSZ * TSZ * HIDSZ * 2;
  unsigned short* wqb  = (unsigned short*)(ws + off); off += (size_t)HIDSZ * HIDSZ * 2;
  unsigned short* wkb  = (unsigned short*)(ws + off); off += (size_t)HIDSZ * HIDSZ * 2;
  unsigned short* wvb  = (unsigned short*)(ws + off); off += (size_t)HIDSZ * HIDSZ * 2;
  unsigned short* dsb  = (unsigned short*)(ws + off); off += (size_t)DROWS * HD * 2;
  unsigned short* Qb   = (unsigned short*)(ws + off); off += (size_t)BSZ * NH * TSZ * HD * 2;
  unsigned short* Kb   = (unsigned short*)(ws + off); off += (size_t)BSZ * NH * TSZ * HD * 2;
  unsigned short* Vt   = (unsigned short*)(ws + off); off += (size_t)BSZ * NH * TSZ * HD * 2;
  float*          wbuf = (float*)(ws + off);          off += (size_t)BSZ * TSZ * 12 * 4;
  (void)ws_size; (void)n_in; (void)in_sizes; (void)out_size;

  cvt_bf16_kernel<<<2048, 256, 0, stream>>>(x,  xb,  BSZ * TSZ * HIDSZ);
  cvt_bf16_kernel<<<1024, 256, 0, stream>>>(Wq, wqb, HIDSZ * HIDSZ);
  cvt_bf16_kernel<<<1024, 256, 0, stream>>>(Wk, wkb, HIDSZ * HIDSZ);
  cvt_bf16_kernel<<<1024, 256, 0, stream>>>(Wv, wvb, HIDSZ * HIDSZ);
  cvt_dist_kernel<<<256, 256, 0, stream>>>(dist, dsb);

  qkv_kernel<<<(BSZ * TSZ) / 16, 128, 0, stream>>>(xb, wqb, wkb, wvb, bq, bk, bv,
                                                   Qb, Kb, Vt);
  gate_kernel<<<(BSZ * TSZ) / 4, 128, 0, stream>>>(x, Wr, br, Wa, wbuf);
  attn_kernel<<<(BSZ * NH * TSZ / 16) / 2, 64, 2 * WAVE_LDS, stream>>>(
      Qb, Kb, Vt, dsb, wbuf, out);
}
